// Decoder_Transformer_70566312673475
// MI455X (gfx1250) — compile-verified, matrix operand order checked by reading, not run on hardware
//
#include <hip/hip_runtime.h>
#include <hip/hip_bf16.h>

// ---------------------------------------------------------------------------
// Problem constants (match the reference)
// ---------------------------------------------------------------------------
constexpr int B_   = 4;
constexpr int S_   = 2048;
constexpr int D_   = 512;
constexpr int H_   = 8;
constexpr int L_   = 4;
constexpr int DFF_ = 2048;
constexpr int DK_  = 64;

typedef __attribute__((ext_vector_type(16))) _Float16 v16h;
typedef __attribute__((ext_vector_type(8)))  _Float16 v8h;
typedef __attribute__((ext_vector_type(8)))  float    v8f;

// ---------------------------------------------------------------------------
// WMMA helpers (CDNA5 16x16x32 f16 -> f32)
// ---------------------------------------------------------------------------
__device__ __forceinline__ v8f wmma_f16(v16h a, v16h b, v8f c) {
  return __builtin_amdgcn_wmma_f32_16x16x32_f16(
      /*neg_a=*/false, a, /*neg_b=*/false, b,
      /*c_mod=*/(short)0, c, /*reuse_a=*/false, /*reuse_b=*/false);
}

// Load a 16x32 f16 A/B fragment from a row-major [16][ld] tile.
// CDNA5 layout: lane = g*16 + r, element e (0..15):
//   k(e) = (e<8 ? g*8 + e : 16 + g*8 + (e-8))  -> two contiguous 8-half runs.
__device__ __forceinline__ v16h load_frag(const _Float16* __restrict__ base, int ld) {
  int lane = threadIdx.x & 31;
  int r = lane & 15;
  int g = lane >> 4;
  const _Float16* p = base + (size_t)r * ld + g * 8;
  v8h lo = *(const v8h*)(p);
  v8h hi = *(const v8h*)(p + 16);
  v16h f;
#pragma unroll
  for (int i = 0; i < 8; ++i) { f[i] = lo[i]; f[8 + i] = hi[i]; }
  return f;
}

// ---------------------------------------------------------------------------
// CDNA5 async memory->LDS copy (ASYNCcnt-tracked); inline asm per
// CDNA5 ISA (GV mode: LDS address taken from the VDST VGPR,
// memory address from the VADDR VGPR). 16 bytes per lane per issue.
// ---------------------------------------------------------------------------
__device__ __forceinline__ void async_b128(unsigned lds_off, const void* gaddr) {
  asm volatile("global_load_async_to_lds_b128 %0, %1, off"
               :: "v"(lds_off), "v"(gaddr)
               : "memory");
}
__device__ __forceinline__ void wait_async0() {
  asm volatile("s_wait_asynccnt 0x0" ::: "memory");
}
__device__ __forceinline__ unsigned lds_off_of(const void* p) {
  // generic pointers into the LDS aperture carry the LDS byte offset in [31:0]
  return (unsigned)(uintptr_t)p;
}

// ---------------------------------------------------------------------------
// fp32 -> fp16 conversion (weights)
// ---------------------------------------------------------------------------
__global__ void __launch_bounds__(256) k_f32_to_f16(const float* __restrict__ a,
                                                    _Float16* __restrict__ o, int n) {
  int i = blockIdx.x * 256 + threadIdx.x;
  if (i < n) o[i] = (_Float16)a[i];
}

// ---------------------------------------------------------------------------
// Embedding: x = src + positional encoding; emit f32 and f16 copies
// ---------------------------------------------------------------------------
__global__ void __launch_bounds__(256) k_embed(const float* __restrict__ src,
                                               float* __restrict__ xf,
                                               _Float16* __restrict__ xh) {
  int idx = blockIdx.x * 256 + threadIdx.x;
  if (idx >= B_ * S_ * D_) return;
  int d  = idx & (D_ - 1);
  int bs = idx >> 9;           // D_=512
  int s  = bs & (S_ - 1);
  float pos = (float)s;
  int   i2  = d >> 1;
  float div = __expf((float)(2 * i2) * (-9.210340371976184f / (float)D_));
  float pe  = (d & 1) ? __cosf(pos * div) : __sinf(pos * div);
  float v   = src[bs] + pe;
  xf[idx] = v;
  xh[idx] = (_Float16)v;
}

// ---------------------------------------------------------------------------
// Tiled WMMA GEMM: C[M,N] = A[M,K] * W[N,K]^T + bias   (A,W f16 row-major)
// Block: 256 threads = 8 waves; tile 128x128; wave tile 32x64; K-step 32.
// Double-buffered LDS filled by CDNA5 async-to-LDS copies (ASYNCcnt).
// ---------------------------------------------------------------------------
template <bool RELU, bool OUTF16>
__global__ void __launch_bounds__(256) k_gemm(
    const _Float16* __restrict__ A,   // [M,K]
    const _Float16* __restrict__ W,   // [N,K]
    const float*    __restrict__ bias,// [N] (may be null)
    float*          __restrict__ Cf,  // [M,N] if !OUTF16
    _Float16*       __restrict__ Ch,  // [M,N] if OUTF16
    int M, int N, int K) {
  __shared__ _Float16 As[2][128 * 32];
  __shared__ _Float16 Bs[2][128 * 32];

  int bm = blockIdx.x, bn = blockIdx.y;
  int tid = threadIdx.x;
  int wid = tid >> 5, lane = tid & 31;
  int wm = wid & 3;          // 4 waves along M (32 rows each)
  int wn = wid >> 2;         // 2 waves along N (64 cols each)

  const _Float16* Ab = A + (size_t)bm * 128 * K;
  const _Float16* Wb = W + (size_t)bn * 128 * K;

  // staging: thread t handles row r = t/2, 16-half chunk c = (t&1)*16 of both tiles
  int r  = tid >> 1;
  int cx = (tid & 1) * 16;
  const _Float16* gA = Ab + (size_t)r * K + cx;
  const _Float16* gW = Wb + (size_t)r * K + cx;
  unsigned la0 = lds_off_of(&As[0][r * 32 + cx]);
  unsigned la1 = lds_off_of(&As[1][r * 32 + cx]);
  unsigned lb0 = lds_off_of(&Bs[0][r * 32 + cx]);
  unsigned lb1 = lds_off_of(&Bs[1][r * 32 + cx]);

  v8f c[2][4] = {};

  // prologue: stage K-tile 0 into buffer 0
  async_b128(la0,      gA);
  async_b128(la0 + 16, gA + 8);
  async_b128(lb0,      gW);
  async_b128(lb0 + 16, gW + 8);
  wait_async0();
  __syncthreads();

  const int nk = K / 32;
  for (int it = 0; it < nk; ++it) {
    int cur = it & 1;
    // prefetch K-tile it+1 into the other buffer (safe: all waves finished
    // reading it at the barrier that ended iteration it-1)
    if (it + 1 < nk) {
      const _Float16* gA2 = gA + (size_t)(it + 1) * 32;
      const _Float16* gW2 = gW + (size_t)(it + 1) * 32;
      unsigned lan = cur ? la0 : la1;
      unsigned lbn = cur ? lb0 : lb1;
      async_b128(lan,      gA2);
      async_b128(lan + 16, gA2 + 8);
      async_b128(lbn,      gW2);
      async_b128(lbn + 16, gW2 + 8);
    }

    const _Float16* Ac = As[cur];
    const _Float16* Bc = Bs[cur];
    v16h a0 = load_frag(Ac + (wm * 32)      * 32, 32);
    v16h a1 = load_frag(Ac + (wm * 32 + 16) * 32, 32);
#pragma unroll
    for (int j = 0; j < 4; ++j) {
      v16h b = load_frag(Bc + (wn * 64 + j * 16) * 32, 32);
      c[0][j] = wmma_f16(a0, b, c[0][j]);
      c[1][j] = wmma_f16(a1, b, c[1][j]);
    }

    wait_async0();      // next tile fully in LDS
    __syncthreads();    // visible to all waves; readers done with cur
  }

  int n_ = lane & 15, g = lane >> 4;
#pragma unroll
  for (int i = 0; i < 2; ++i)
#pragma unroll
    for (int j = 0; j < 4; ++j) {
      int row0 = bm * 128 + wm * 32 + i * 16;
      int col  = bn * 128 + wn * 64 + j * 16 + n_;
      float bv = bias ? bias[col] : 0.0f;
#pragma unroll
      for (int rr = 0; rr < 8; ++rr) {
        float v = c[i][j][rr] + bv;
        if (RELU) v = fmaxf(v, 0.0f);
        int row = row0 + g * 8 + rr;
        if (OUTF16) Ch[(size_t)row * N + col] = (_Float16)v;
        else        Cf[(size_t)row * N + col] = v;
      }
    }
}

// ---------------------------------------------------------------------------
// V transpose: [B,S,D] f16 -> Vt [B,H,DK,S] f16 (so P*V B-fragments are
// contiguous row-major loads)
// ---------------------------------------------------------------------------
__global__ void __launch_bounds__(256) k_transpose_v(const _Float16* __restrict__ v,
                                                     _Float16* __restrict__ vt) {
  int idx = blockIdx.x * 256 + threadIdx.x;
  if (idx >= B_ * S_ * D_) return;
  int d  = idx & (D_ - 1);
  int bs = idx >> 9;
  int s  = bs & (S_ - 1);
  int b  = bs >> 11;
  int h  = d >> 6;
  int dd = d & (DK_ - 1);
  vt[(((size_t)(b * H_ + h) * DK_) + dd) * S_ + s] = v[idx];
}

// ---------------------------------------------------------------------------
// Attention: one workgroup per (b, h, 16-query block).
//  phase1: Sc[16][S] = mask(QK^T * scale)   (WMMA)
//  phase2: row softmax in LDS
//  phase3: O = P * V                        (WMMA, Vt pre-transposed)
// Dynamic LDS: Sc 128KB + Qt 2KB + Pt 2KB + Red 2KB + Acc 4KB = 141312 B
// ---------------------------------------------------------------------------
__global__ void __launch_bounds__(256) k_attention(
    const _Float16* __restrict__ Q,   // [B,S,D]
    const _Float16* __restrict__ Km,  // [B,S,D]
    const _Float16* __restrict__ Vt,  // [B,H,DK,S]
    const float*    __restrict__ src, // [B,S,1] (mask source)
    _Float16*       __restrict__ O) { // [B,S,D]
  extern __shared__ char smem[];
  float*    Sc  = (float*)smem;                      // [16][S_]
  _Float16* Qt  = (_Float16*)(smem + 16 * S_ * 4);   // [16][64]
  _Float16* Pt  = Qt + 16 * 64;                      // [16][64]
  float*    Red = (float*)(smem + 16 * S_ * 4 + 4096);  // [2][16][16]
  float*    Acc = Red + 512;                         // [4][16][16]

  int wg = blockIdx.x;
  int qb = wg & (S_ / 16 - 1);
  int t  = wg >> 7;                  // / (S_/16)=128
  int h  = t & (H_ - 1);
  int b  = t >> 3;

  int tid = threadIdx.x, wid = tid >> 5, lane = tid & 31;
  int n_ = lane & 15, g = lane >> 4;

  const _Float16* Qbase = Q + ((size_t)b * S_ + qb * 16) * D_ + h * DK_;
  for (int i = tid; i < 16 * DK_; i += 256) {
    int r = i >> 6, cx = i & 63;
    Qt[i] = Qbase[(size_t)r * D_ + cx];
  }
  __syncthreads();

  // ---- phase 1: scores ----
  const float scale = 0.125f;   // 1/sqrt(64)
  const _Float16* Kbase = Km + (size_t)b * S_ * D_ + h * DK_;
  v16h qa0 = load_frag(Qt, DK_);        // k = 0..31
  v16h qa1 = load_frag(Qt + 32, DK_);   // k = 32..63
  for (int sb = 0; sb < S_; sb += 128) {
    int key0 = sb + wid * 16;
    v8f c = {};
    v16h kb0 = load_frag(Kbase + (size_t)key0 * D_, D_);
    v16h kb1 = load_frag(Kbase + (size_t)key0 * D_ + 32, D_);
    c = wmma_f16(qa0, kb0, c);
    c = wmma_f16(qa1, kb1, c);
    int key = key0 + n_;
    bool valid = (src[(size_t)b * S_ + key] != 0.0f);
#pragma unroll
    for (int r = 0; r < 8; ++r) {
      float v = c[r] * scale;
      Sc[(size_t)(g * 8 + r) * S_ + key] = valid ? v : -1e9f;
    }
  }
  __syncthreads();

  // ---- phase 2: softmax over each of 16 rows (16 threads per row) ----
  {
    int row = tid >> 4, sub = tid & 15;
    float* R0 = Red;
    float* R1 = Red + 256;
    float mx = -3.4e38f;
    for (int cx = sub; cx < S_; cx += 16) mx = fmaxf(mx, Sc[(size_t)row * S_ + cx]);
    R0[row * 16 + sub] = mx;
    __syncthreads();
    mx = R0[row * 16];
#pragma unroll
    for (int i = 1; i < 16; ++i) mx = fmaxf(mx, R0[row * 16 + i]);
    float sm = 0.0f;
    for (int cx = sub; cx < S_; cx += 16) {
      float e = __expf(Sc[(size_t)row * S_ + cx] - mx);
      Sc[(size_t)row * S_ + cx] = e;
      sm += e;
    }
    R1[row * 16 + sub] = sm;
    __syncthreads();
    sm = 0.0f;
#pragma unroll
    for (int i = 0; i < 16; ++i) sm += R1[row * 16 + i];
    float inv = 1.0f / sm;
    for (int cx = sub; cx < S_; cx += 16) Sc[(size_t)row * S_ + cx] *= inv;
  }

  // ---- phase 3: O = P * V ----
  int ntile = wid & 3;   // which 16-wide d-column tile
  int khalf = wid >> 2;  // split keys within each 64-chunk
  const _Float16* Vbase = Vt + (size_t)(b * H_ + h) * DK_ * S_;  // [64][S_]
  v8f acc = {};
  for (int kc = 0; kc < S_; kc += 64) {
    __syncthreads();   // also covers phase-2 completion on first pass
    for (int i = tid; i < 16 * 64; i += 256) {
      int r = i >> 6, cx = i & 63;
      Pt[i] = (_Float16)Sc[(size_t)r * S_ + kc + cx];
    }
    __syncthreads();
    v16h pa = load_frag(Pt + khalf * 32, 64);
    v16h vb = load_frag(Vbase + (size_t)(ntile * 16) * S_ + kc + khalf * 32, S_);
    acc = wmma_f16(pa, vb, acc);
  }
  __syncthreads();
  if (khalf == 1) {
#pragma unroll
    for (int r = 0; r < 8; ++r)
      Acc[ntile * 256 + (g * 8 + r) * 16 + n_] = acc[r];
  }
  __syncthreads();
  if (khalf == 0) {
    _Float16* Ob = O + ((size_t)b * S_ + qb * 16) * D_ + h * DK_ + ntile * 16;
#pragma unroll
    for (int r = 0; r < 8; ++r) {
      float v = acc[r] + Acc[ntile * 256 + (g * 8 + r) * 16 + n_];
      Ob[(size_t)(g * 8 + r) * D_ + n_] = (_Float16)v;
    }
  }
}

// ---------------------------------------------------------------------------
// Residual + LayerNorm (row of 512); writes f32 and f16 x
// ---------------------------------------------------------------------------
__global__ void __launch_bounds__(128) k_add_ln(
    const float* __restrict__ x, const float* __restrict__ y,
    const float* __restrict__ gg, const float* __restrict__ bb,
    float* __restrict__ xo, _Float16* __restrict__ xo16) {
  __shared__ float red[128];
  int row = blockIdx.x, tid = threadIdx.x;
  const float* xr = x + (size_t)row * D_;
  const float* yr = y + (size_t)row * D_;
  float vals[4];
  float s = 0.0f;
#pragma unroll
  for (int i = 0; i < 4; ++i) {
    float v = xr[tid + 128 * i] + yr[tid + 128 * i];
    vals[i] = v;
    s += v;
  }
  red[tid] = s;
  __syncthreads();
  for (int off = 64; off > 0; off >>= 1) {
    if (tid < off) red[tid] += red[tid + off];
    __syncthreads();
  }
  float mean = red[0] / (float)D_;
  __syncthreads();
  float s2 = 0.0f;
#pragma unroll
  for (int i = 0; i < 4; ++i) { float d = vals[i] - mean; s2 += d * d; }
  red[tid] = s2;
  __syncthreads();
  for (int off = 64; off > 0; off >>= 1) {
    if (tid < off) red[tid] += red[tid + off];
    __syncthreads();
  }
  float inv = rsqrtf(red[0] / (float)D_ + 1e-5f);
#pragma unroll
  for (int i = 0; i < 4; ++i) {
    int cx = tid + 128 * i;
    float v = (vals[i] - mean) * inv * gg[cx] + bb[cx];
    xo[(size_t)row * D_ + cx]   = v;
    xo16[(size_t)row * D_ + cx] = (_Float16)v;
  }
}

// ---------------------------------------------------------------------------
// Final projection: out[row] = dot(x[row,:], Wf) + bf
// ---------------------------------------------------------------------------
__global__ void __launch_bounds__(128) k_final(const float* __restrict__ x,
                                               const float* __restrict__ wf,
                                               const float* __restrict__ bf,
                                               float* __restrict__ out) {
  __shared__ float red[128];
  int row = blockIdx.x, tid = threadIdx.x;
  float s = 0.0f;
  for (int i = tid; i < D_; i += 128) s += x[(size_t)row * D_ + i] * wf[i];
  red[tid] = s;
  __syncthreads();
  for (int off = 64; off > 0; off >>= 1) {
    if (tid < off) red[tid] += red[tid + off];
    __syncthreads();
  }
  if (tid == 0) out[row] = red[0] + bf[0];
}

// ---------------------------------------------------------------------------
// Host orchestration
// ---------------------------------------------------------------------------
extern "C" void kernel_launch(void* const* d_in, const int* in_sizes, int n_in,
                              void* d_out, int out_size, void* d_ws, size_t ws_size,
                              hipStream_t stream) {
  const float* src = (const float*)d_in[0];
  const float* Wq  = (const float*)d_in[1];  const float* bq = (const float*)d_in[2];
  const float* Wk  = (const float*)d_in[3];  const float* bk = (const float*)d_in[4];
  const float* Wv  = (const float*)d_in[5];  const float* bv = (const float*)d_in[6];
  const float* Wo  = (const float*)d_in[7];  const float* bo = (const float*)d_in[8];
  const float* W1  = (const float*)d_in[9];  const float* b1 = (const float*)d_in[10];
  const float* W2  = (const float*)d_in[11]; const float* b2 = (const float*)d_in[12];
  const float* g1  = (const float*)d_in[13]; const float* be1 = (const float*)d_in[14];
  const float* g2  = (const float*)d_in[15]; const float* be2 = (const float*)d_in[16];
  const float* Wf  = (const float*)d_in[17]; const float* bf = (const float*)d_in[18];
  float* out = (float*)d_out;

  size_t off = 0;
  auto wsalloc = [&](size_t bytes) -> void* {
    void* p = (char*)d_ws + off;
    off += (bytes + 255) & ~(size_t)255;
    return p;
  };
  const size_t DD  = (size_t)D_ * D_;
  const size_t DFD = (size_t)DFF_ * D_;
  const size_t BSD = (size_t)B_ * S_ * D_;
  const size_t BSF = (size_t)B_ * S_ * DFF_;

  _Float16* wq16 = (_Float16*)wsalloc(L_ * DD * 2);
  _Float16* wk16 = (_Float16*)wsalloc(L_ * DD * 2);
  _Float16* wv16 = (_Float16*)wsalloc(L_ * DD * 2);
  _Float16* wo16 = (_Float16*)wsalloc(L_ * DD * 2);
  _Float16* w116 = (_Float16*)wsalloc(L_ * DFD * 2);
  _Float16* w216 = (_Float16*)wsalloc(L_ * DFD * 2);
  float*    xf   = (float*)   wsalloc(BSD * 4);
  _Float16* xh   = (_Float16*)wsalloc(BSD * 2);
  _Float16* q16  = (_Float16*)wsalloc(BSD * 2);
  _Float16* k16  = (_Float16*)wsalloc(BSD * 2);
  _Float16* v16  = (_Float16*)wsalloc(BSD * 2);
  _Float16* vt16 = (_Float16*)wsalloc(BSD * 2);
  _Float16* ao16 = (_Float16*)wsalloc(BSD * 2);
  float*    yf   = (float*)   wsalloc(BSD * 4);
  _Float16* h16  = (_Float16*)wsalloc(BSF * 2);
  (void)ws_size; (void)n_in; (void)in_sizes; (void)out_size;

  // weight conversion f32 -> f16
  {
    int n1 = (int)(L_ * DD);
    k_f32_to_f16<<<(n1 + 255) / 256, 256, 0, stream>>>(Wq, wq16, n1);
    k_f32_to_f16<<<(n1 + 255) / 256, 256, 0, stream>>>(Wk, wk16, n1);
    k_f32_to_f16<<<(n1 + 255) / 256, 256, 0, stream>>>(Wv, wv16, n1);
    k_f32_to_f16<<<(n1 + 255) / 256, 256, 0, stream>>>(Wo, wo16, n1);
    int n2 = (int)(L_ * DFD);
    k_f32_to_f16<<<(n2 + 255) / 256, 256, 0, stream>>>(W1, w116, n2);
    k_f32_to_f16<<<(n2 + 255) / 256, 256, 0, stream>>>(W2, w216, n2);
  }

  // embedding
  k_embed<<<(int)((BSD + 255) / 256), 256, 0, stream>>>(src, xf, xh);

  const int M = B_ * S_;                 // 8192
  dim3 gD(M / 128, D_ / 128);            // N = 512  -> (64, 4)
  dim3 gF(M / 128, DFF_ / 128);          // N = 2048 -> (64, 16)
  const size_t attn_smem = (size_t)16 * S_ * 4 + 4096 + 2048 + 4096; // 141312 B

  for (int l = 0; l < L_; ++l) {
    const _Float16* wql = wq16 + (size_t)l * DD;
    const _Float16* wkl = wk16 + (size_t)l * DD;
    const _Float16* wvl = wv16 + (size_t)l * DD;
    const _Float16* wol = wo16 + (size_t)l * DD;
    const _Float16* w1l = w116 + (size_t)l * DFD;
    const _Float16* w2l = w216 + (size_t)l * DFD;

    // Q, K, V projections (f16 out)
    k_gemm<false, true><<<gD, 256, 0, stream>>>(xh, wql, bq + l * D_, nullptr, q16, M, D_, D_);
    k_gemm<false, true><<<gD, 256, 0, stream>>>(xh, wkl, bk + l * D_, nullptr, k16, M, D_, D_);
    k_gemm<false, true><<<gD, 256, 0, stream>>>(xh, wvl, bv + l * D_, nullptr, v16, M, D_, D_);

    k_transpose_v<<<(int)((BSD + 255) / 256), 256, 0, stream>>>(v16, vt16);

    // attention
    k_attention<<<B_ * H_ * (S_ / 16), 256, attn_smem, stream>>>(q16, k16, vt16, src, ao16);

    // output projection (f32 out) + residual layernorm
    k_gemm<false, false><<<gD, 256, 0, stream>>>(ao16, wol, bo + l * D_, yf, nullptr, M, D_, D_);
    k_add_ln<<<M, 128, 0, stream>>>(xf, yf, g1 + l * D_, be1 + l * D_, xf, xh);

    // FFN
    k_gemm<true,  true ><<<gF, 256, 0, stream>>>(xh, w1l, b1 + l * DFF_, nullptr, h16, M, DFF_, D_);
    k_gemm<false, false><<<gD, 256, 0, stream>>>(h16, w2l, b2 + l * D_, yf, nullptr, M, D_, DFF_);
    k_add_ln<<<M, 128, 0, stream>>>(xf, yf, g2 + l * D_, be2 + l * D_, xf, xh);
  }

  // final projection [B,S,1]
  k_final<<<M, 128, 0, stream>>>(xf, Wf, bf, out);
}
